// MoESharedProjectionHead_44607530336415
// MI455X (gfx1250) — matrix operand order repaired and checked by reading, not compiled, not run on hardware
//
#include <hip/hip_runtime.h>

// CDNA5 / gfx1250 fused MoE projection head, v3.
// Per 64-position tile: h = ReLU(BNfold(W_pre)@x) via v_wmma_f32_16x16x32_bf16
// into LDS; router logits via WMMA (softmax skipped: monotonic for top-k);
// per-expert COLUMN COMPACTION (LDS atomics build per-expert column lists) so
// expert GEMMs run only on gathered active columns (~4x less WMMA work than
// dense+mask); group-sum accumulated in LDS, coalesced f32 store.
// v3: fragment loads hoisted ahead of the WMMA groups so the backend can
// clause the ds_load_b128s, wait once, and issue WMMAs back-to-back.

typedef __attribute__((ext_vector_type(16))) __bf16 v16bf;
typedef __attribute__((ext_vector_type(8)))  float  v8f;

#define NT 64  // positions per tile

__device__ __forceinline__ v8f wmma_bf16(v16bf a, v16bf b, v8f c) {
  return __builtin_amdgcn_wmma_f32_16x16x32_bf16(false, a, false, b, (short)0, c,
                                                 false, false);
}

// A fragment: 16x32 (MxK) bf16, row-major source, stride lda.
// lane m = l&15; lanes 0-15 hold K 0-7 & 16-23, lanes 16-31 hold K 8-15 & 24-31.
__device__ __forceinline__ v16bf load_a_frag(const __bf16* __restrict__ A,
                                             int lda, int lane) {
  const __bf16* row = A + (lane & 15) * lda + ((lane >> 4) << 3);
  v16bf f;
#pragma unroll
  for (int v = 0; v < 8; ++v) {
    int kb = ((v & 4) << 2) + ((v & 3) << 1);  // v<4: 0,2,4,6  v>=4: 16,18,20,22
    f[2 * v]     = row[kb];
    f[2 * v + 1] = row[kb + 1];
  }
  return f;
}

// B fragment: 32x16 (KxN) bf16 from LDS stored position-major (bt[pos*ld + k]).
// Per-lane column index 'col' (lanes 0-15 hold K 0-7 & 16-23, lanes 16-31 hold
// K 8-15 & 24-31 of the same column); two 8-bf16 chunks -> ds_load_b128 pairs.
__device__ __forceinline__ v16bf gather_b_frag(const __bf16* bt, int ld, int k0,
                                               int col, int lane) {
  const __bf16* p = bt + col * ld + k0 + ((lane >> 4) << 3);
  v16bf f;
#pragma unroll
  for (int v = 0; v < 16; ++v) f[v] = p[v + ((v >> 3) << 3)];
  return f;
}

__device__ __forceinline__ v16bf load_b_frag(const __bf16* bt, int ld, int k0,
                                             int n0, int lane) {
  return gather_b_frag(bt, ld, k0, n0 + (lane & 15), lane);
}

// ---------------- prep: fold BN into W_pre/bias, convert weights to bf16 ----
__global__ void prep_weights(const float* __restrict__ w_pre,
                             const float* __restrict__ b_pre,
                             const float* __restrict__ gamma,
                             const float* __restrict__ beta,
                             const float* __restrict__ mean,
                             const float* __restrict__ var,
                             const float* __restrict__ w_r,
                             const float* __restrict__ w_exp,
                             __bf16* __restrict__ Wp, float* __restrict__ bp,
                             __bf16* __restrict__ wrb, __bf16* __restrict__ web,
                             int C) {
  const int n_wp = C * C;
  const int n_we = 2048 * C;  // 16 experts * 128 * C
  const int n_wr = 16 * C;
  const int total = n_wp + n_we + n_wr + C;
  for (int i = blockIdx.x * blockDim.x + threadIdx.x; i < total;
       i += gridDim.x * blockDim.x) {
    if (i < n_wp) {
      int o = i / C;
      float inv = rsqrtf(var[o] + 1e-5f);
      Wp[i] = (__bf16)(w_pre[i] * gamma[o] * inv);
    } else if (i < n_wp + n_we) {
      web[i - n_wp] = (__bf16)w_exp[i - n_wp];
    } else if (i < n_wp + n_we + n_wr) {
      wrb[i - n_wp - n_we] = (__bf16)w_r[i - n_wp - n_we];
    } else {
      int o = i - n_wp - n_we - n_wr;
      float inv = rsqrtf(var[o] + 1e-5f);
      bp[o] = b_pre[o] * gamma[o] * inv + beta[o] - gamma[o] * mean[o] * inv;
    }
  }
}

// ---------------- fused per-scale kernel ------------------------------------
template <int C>
__global__ void __launch_bounds__(256)
moe_fused(const float* __restrict__ x, const __bf16* __restrict__ Wp,
          const float* __restrict__ bp, const __bf16* __restrict__ wr,
          const float* __restrict__ br, const __bf16* __restrict__ wexp,
          const float* __restrict__ bexp, float* __restrict__ out, int HW) {
  constexpr int LDX = C + 8;                    // bf16 row stride
  constexpr int XS_BYTES  = NT * LDX * 2;       // x/h tile bytes
  constexpr int OA_BYTES  = 256 * (NT + 1) * 4; // out accumulator bytes
  constexpr int R0_BYTES  = (XS_BYTES > OA_BYTES) ? XS_BYTES : OA_BYTES;

  extern __shared__ char smem[];
  __bf16* xs      = (__bf16*)smem;                    // phase 0/1 (dies after 1)
  float*  out_acc = (float*)smem;                     // union with xs
  __bf16* hs      = (__bf16*)(smem + R0_BYTES);       // [NT][LDX] h tile
  float*  logits  = (float*)(smem + R0_BYTES + XS_BYTES);  // [NT][16]
  int*    list    = (int*)(logits + NT * 16);         // [16][NT] column lists
  int*    cnt     = list + 16 * NT;                   // [16]

  const int tid  = threadIdx.x;
  const int lane = tid & 31;
  const int wave = tid >> 5;
  const int ntile  = (HW + NT - 1) / NT;
  const int bidx   = blockIdx.x / ntile;
  const int p0     = (blockIdx.x % ntile) * NT;
  const int nvalid = min(NT, HW - p0);

  // ---- Phase 0: stage x tile (f32 -> bf16, transposed to position-major)
  {
    const float* xb = x + (size_t)bidx * C * HW + p0;
    const int n = tid & 63;
    for (int c = tid >> 6; c < C; c += 4)
      xs[n * LDX + c] = (__bf16)((n < nvalid) ? xb[c * HW + n] : 0.f);
  }
  __syncthreads();

  // ---- Phase 1: h = ReLU(W' @ xtile + b')   (M=C, N=64, K=C)
  for (int mt = wave * 16; mt < C; mt += 128) {
    v8f a0 = {}, a1 = {}, a2 = {}, a3 = {};
#pragma unroll
    for (int kk = 0; kk < C; kk += 32) {
      // hoist all fragment loads -> one ds clause + single wait, then 4 WMMAs
      v16bf a  = load_a_frag(Wp + mt * C + kk, C, lane);
      v16bf b0 = load_b_frag(xs, LDX, kk, 0,  lane);
      v16bf b1 = load_b_frag(xs, LDX, kk, 16, lane);
      v16bf b2 = load_b_frag(xs, LDX, kk, 32, lane);
      v16bf b3 = load_b_frag(xs, LDX, kk, 48, lane);
      a0 = wmma_bf16(a, b0, a0);
      a1 = wmma_bf16(a, b1, a1);
      a2 = wmma_bf16(a, b2, a2);
      a3 = wmma_bf16(a, b3, a3);
    }
    const int n = lane & 15, hi = lane >> 4;
#pragma unroll
    for (int v = 0; v < 8; ++v) {
      int m = mt + v + 8 * hi;  // C/D layout: VGPR v -> M = v + 8*hi
      float bm = bp[m];
      hs[(n +  0) * LDX + m] = (__bf16)fmaxf(a0[v] + bm, 0.f);
      hs[(n + 16) * LDX + m] = (__bf16)fmaxf(a1[v] + bm, 0.f);
      hs[(n + 32) * LDX + m] = (__bf16)fmaxf(a2[v] + bm, 0.f);
      hs[(n + 48) * LDX + m] = (__bf16)fmaxf(a3[v] + bm, 0.f);
    }
  }
  __syncthreads();  // xs dead from here; out_acc takes over the region

  // ---- Phase 2a: wave0 computes router logits (16xNT) via WMMA;
  //      waves 1..7 zero out_acc, wave1 zeros expert counters.
  if (wave == 0) {
    v8f r0 = {}, r1 = {}, r2 = {}, r3 = {};
#pragma unroll
    for (int kk = 0; kk < C; kk += 32) {
      v16bf a  = load_a_frag(wr + kk, C, lane);  // 16 router rows exactly
      v16bf b0 = load_b_frag(hs, LDX, kk, 0,  lane);
      v16bf b1 = load_b_frag(hs, LDX, kk, 16, lane);
      v16bf b2 = load_b_frag(hs, LDX, kk, 32, lane);
      v16bf b3 = load_b_frag(hs, LDX, kk, 48, lane);
      r0 = wmma_bf16(a, b0, r0);
      r1 = wmma_bf16(a, b1, r1);
      r2 = wmma_bf16(a, b2, r2);
      r3 = wmma_bf16(a, b3, r3);
    }
    const int n = lane & 15, hi = lane >> 4;
#pragma unroll
    for (int v = 0; v < 8; ++v) {
      int e = v + 8 * hi;
      float be = br[e];
      logits[(n +  0) * 16 + e] = r0[v] + be;
      logits[(n + 16) * 16 + e] = r1[v] + be;
      logits[(n + 32) * 16 + e] = r2[v] + be;
      logits[(n + 48) * 16 + e] = r3[v] + be;
    }
  } else {
    for (int i = tid - 32; i < 256 * (NT + 1); i += 224) out_acc[i] = 0.f;
    if (tid >= 32 && tid < 48) cnt[tid - 32] = 0;
  }
  __syncthreads();

  // ---- Phase 2b: top-2 per column (ties: first index, as in lax.top_k),
  //      append column to its two experts' lists (LDS ds_add_rtn atomics).
  if (tid < nvalid) {
    const float* lg = logits + tid * 16;
    float b1v = -3.0e38f; int i1 = 0;
#pragma unroll
    for (int e = 0; e < 16; ++e)
      if (lg[e] > b1v) { b1v = lg[e]; i1 = e; }
    float b2v = -3.0e38f; int i2 = 0;
#pragma unroll
    for (int e = 0; e < 16; ++e)
      if (e != i1 && lg[e] > b2v) { b2v = lg[e]; i2 = e; }
    int s1 = atomicAdd(&cnt[i1], 1); list[i1 * NT + s1] = tid;
    int s2 = atomicAdd(&cnt[i2], 1); list[i2 * NT + s2] = tid;
  }
  __syncthreads();

  // ---- Phase 3: compacted expert GEMMs. For expert e only ceil(ne/16)
  //      gathered 16-column fragments are computed (vs 4 dense), then
  //      scatter-accumulated into the group-sum with the expert bias.
  for (int e = 0; e < 16; ++e) {
    const int ne = __builtin_amdgcn_readfirstlane(cnt[e]);  // uniform
    if (ne == 0) continue;
    if (e + 1 < 16)
      __builtin_prefetch(wexp + (size_t)(e + 1) * 128 * C, 0, 1);

    const __bf16* We = wexp + (size_t)e * 128 * C;
    const int mb = wave * 16;  // 8 waves x 16 rows = 128 expert channels
    const int s  = e & 1;      // group slot: out rows [s*128, s*128+128)
    for (int g = 0; g < ne; g += 16) {
      const int n = lane & 15;
      const bool valid = (g + n) < ne;
      const int col = list[e * NT + (valid ? (g + n) : 0)];
      v8f acc = {};
#pragma unroll
      for (int kk = 0; kk < C; kk += 32) {
        v16bf a = load_a_frag(We + mb * C + kk, C, lane);
        v16bf b = gather_b_frag(hs, LDX, kk, col, lane);
        acc = wmma_bf16(a, b, acc);
      }
      const int hi = lane >> 4;
#pragma unroll
      for (int v = 0; v < 8; ++v) {
        int m = mb + v + 8 * hi;
        if (valid)
          out_acc[(s * 128 + m) * (NT + 1) + col] += acc[v] + bexp[e * 128 + m];
      }
    }
  }
  __syncthreads();

  // ---- Phase 4: coalesced store of the 256 x nvalid output tile
  {
    const int n = tid & 63;
    float* ob = out + (size_t)bidx * 256 * HW + p0;
    if (n < nvalid)
      for (int r = tid >> 6; r < 256; r += 4)
        ob[(size_t)r * HW + n] = out_acc[r * (NT + 1) + n];
  }
}

// ---------------- host launcher ---------------------------------------------
// Input flattening assumed (setup_inputs order, params dict in insertion
// order): [0..2]=x0,x1,x2 then per scale s0,s1,s2:
// w_pre,b_pre,bn_gamma,bn_beta,bn_mean,bn_var,w_r,b_r,w_exp,b_exp.
extern "C" void kernel_launch(void* const* d_in, const int* in_sizes, int n_in,
                              void* d_out, int out_size, void* d_ws,
                              size_t ws_size, hipStream_t stream) {
  (void)in_sizes; (void)n_in; (void)out_size; (void)ws_size;
  const float* xin[3] = {(const float*)d_in[0], (const float*)d_in[1],
                         (const float*)d_in[2]};
  const int Cs[3]  = {128, 256, 512};
  const int HWs[3] = {6400, 1600, 400};
  const int PB[3]  = {3, 13, 23};

  char* ws = (char*)d_ws;
  size_t off = 0;
  auto take = [&](size_t bytes) -> void* {
    void* p = ws + off;
    off += (bytes + 255) & ~(size_t)255;
    return p;
  };

  __bf16* Wp[3]; float* bp[3]; __bf16* wrb[3]; __bf16* web[3];
  for (int s = 0; s < 3; ++s) {
    size_t C = (size_t)Cs[s];
    Wp[s]  = (__bf16*)take(C * C * 2);
    bp[s]  = (float*)take(C * 4);
    wrb[s] = (__bf16*)take(16 * C * 2);
    web[s] = (__bf16*)take(2048 * C * 2);
  }

  for (int s = 0; s < 3; ++s) {
    const int pb = PB[s];
    prep_weights<<<dim3(1024), dim3(256), 0, stream>>>(
        (const float*)d_in[pb + 0], (const float*)d_in[pb + 1],
        (const float*)d_in[pb + 2], (const float*)d_in[pb + 3],
        (const float*)d_in[pb + 4], (const float*)d_in[pb + 5],
        (const float*)d_in[pb + 6], (const float*)d_in[pb + 8], Wp[s], bp[s],
        wrb[s], web[s], Cs[s]);
  }

  float* out0 = (float*)d_out;
  float* out1 = out0 + (size_t)4 * 256 * 6400;
  float* out2 = out1 + (size_t)4 * 256 * 1600;

  auto shbytes = [](int C) -> size_t {
    size_t xsb = (size_t)NT * (C + 8) * 2;
    size_t oab = (size_t)256 * (NT + 1) * 4;
    size_t r0 = xsb > oab ? xsb : oab;
    return r0 + xsb + (size_t)NT * 16 * 4 + (size_t)16 * NT * 4 + 64;
  };
  auto nblk = [](int HW) { return 4 * ((HW + NT - 1) / NT); };

  moe_fused<128><<<dim3(nblk(HWs[0])), dim3(256), shbytes(128), stream>>>(
      xin[0], Wp[0], bp[0], wrb[0], (const float*)d_in[PB[0] + 7], web[0],
      (const float*)d_in[PB[0] + 9], out0, HWs[0]);
  moe_fused<256><<<dim3(nblk(HWs[1])), dim3(256), shbytes(256), stream>>>(
      xin[1], Wp[1], bp[1], wrb[1], (const float*)d_in[PB[1] + 7], web[1],
      (const float*)d_in[PB[1] + 9], out1, HWs[1]);
  moe_fused<512><<<dim3(nblk(HWs[2])), dim3(256), shbytes(512), stream>>>(
      xin[2], Wp[2], bp[2], wrb[2], (const float*)d_in[PB[2] + 7], web[2],
      (const float*)d_in[PB[2] + 9], out2, HWs[2]);
}
